// Agent_70428873720081
// MI455X (gfx1250) — matrix operand order repaired and checked by reading, not compiled, not run on hardware
//
#include <hip/hip_runtime.h>
#include <hip/hip_bf16.h>

// ---------------------------------------------------------------------------
// GNN forward: GCNConv -> BatchNorm+ReLU -> 3-layer MLP head.
// MI455X strategy:
//  * GCN GEMM (1024x29 @ 29x128) -> single v_wmma_f32_16x16x32_f16 per tile
//    (K=29 padded to 32, f16 inputs, f32 accumulate; BN renormalizes after).
//  * Edge aggregation via global f32 atomics (L2-resident, ~17MB traffic).
//  * MLP GEMVs are pure HBM streaming (805MB of f32 weights, used once):
//    coalesced float4 loads with non-temporal hints, split-K + atomics for
//    the 131072-deep reduction. Roofline ~35us @ 23.3 TB/s.
// ---------------------------------------------------------------------------

#define NN   1024
#define EE   32768
#define FIN  29
#define HH   128
#define EH   256
#define MAXE 523776           // N*(N-1)/2

// workspace layout (float offsets)
#define OFF_HW   0            // 131072 : h = x@W (later reused as BN output v)
#define OFF_AGG  131072       // 131072 : scatter-add accumulator
#define OFF_DEG  262144       // 1024   : degree -> dinv (in place)
#define OFF_Y1   263168       // 512    : layer-1 split-K accumulator
#define OFF_Y2   263680       // 256    : layer-2 output
#define OFF_MEAN 263936       // 128
#define OFF_SCL  264064       // 128
#define ZERO_CNT 132608       // agg + deg + y1 are contiguous and atomically accumulated

typedef __attribute__((ext_vector_type(16))) _Float16 v16h;
typedef __attribute__((ext_vector_type(8)))  float    v8f;
typedef __attribute__((ext_vector_type(4)))  float    v4f;

// ---------------- zero the accumulated regions (graph-replay safe) ----------
__global__ void k_zero(float* __restrict__ p, int n) {
    int i = blockIdx.x * blockDim.x + threadIdx.x;
    if (i < n) p[i] = 0.0f;
}

// ---------------- GCN GEMM via WMMA: hw[1024][128] = x[1024][29] @ W[29][128]
// One wave per 16x16 output tile. 64 M-tiles x 8 N-tiles = 512 waves.
__global__ __launch_bounds__(256) void k_gcn_gemm(const float* __restrict__ x,
                                                  const float* __restrict__ W,
                                                  float* __restrict__ hw) {
    const int lane   = threadIdx.x & 31;
    const int wave   = threadIdx.x >> 5;
    const int tile   = blockIdx.x * 8 + wave;      // 0..511
    const int tm     = tile >> 3;                  // 0..63
    const int tn     = tile & 7;                   // 0..7
    const int hi     = lane >> 4;                  // 0: lanes 0-15, 1: lanes 16-31
    const int l16    = lane & 15;

    // A: 16x32 f16, lane<16 holds K=0..15 ; lane>=16 holds K=16..31
    v16h a;
    const int m   = tm * 16 + l16;
    const int kbA = hi ? 16 : 0;
#pragma unroll
    for (int h = 0; h < 16; ++h) {
        int k = kbA + h;
        a[h] = (k < FIN) ? (_Float16)x[m * FIN + k] : (_Float16)0.0f;
    }

    // B: 32x16 f16, VGPR g holds K={kb+2g, kb+2g+1}; kb = 0 (lanes 0-15) / 16
    v16h b;
    const int n   = tn * 16 + l16;
    const int kbB = hi ? 16 : 0;
#pragma unroll
    for (int h = 0; h < 16; ++h) {
        int k = kbB + h;
        b[h] = (k < FIN) ? (_Float16)W[k * HH + n] : (_Float16)0.0f;
    }

    v8f c = {};
    c = __builtin_amdgcn_wmma_f32_16x16x32_f16(false, a, false, b,
                                               (short)0, c, false, false);

    // C/D: VGPR g, lanes 0-15 -> M=g, lanes 16-31 -> M=8+g ; N=lane&15
#pragma unroll
    for (int g = 0; g < 8; ++g) {
        int row = tm * 16 + g + hi * 8;
        hw[row * HH + n] = c[g];
    }
}

// ---------------- degree (with self loops), then dinv ----------------------
__global__ void k_deg(const int* __restrict__ ei, float* __restrict__ deg) {
    int t = blockIdx.x * blockDim.x + threadIdx.x;
    if (t < EE)              atomicAdd(&deg[ei[EE + t]], 1.0f);   // dst of edge
    else if (t < EE + NN)    atomicAdd(&deg[t - EE], 1.0f);       // self loop
}

__global__ void k_dinv(float* __restrict__ deg) {
    int i = blockIdx.x * blockDim.x + threadIdx.x;
    if (i < NN) {
        float d = deg[i];
        deg[i] = (d > 0.0f) ? rsqrtf(d) : 0.0f;
    }
}

// ---------------- message scatter: agg[dst] += dinv[s]*dinv[d]*hw[src] ------
__global__ void k_scatter(const int* __restrict__ ei,
                          const float* __restrict__ hw,
                          const float* __restrict__ dinv,
                          float* __restrict__ agg) {
    int t = blockIdx.x * blockDim.x + threadIdx.x;     // (E+N)*128 threads
    int e = t >> 7;
    int c = t & 127;
    if (e >= EE + NN) return;
    int s, d;
    if (e < EE) { s = ei[e]; d = ei[EE + e]; }
    else        { s = d = e - EE; }
    float nrm = dinv[s] * dinv[d];
    atomicAdd(&agg[d * HH + c], nrm * hw[s * HH + c]);
}

// ---------------- BN statistics: one block, one channel per thread ----------
__global__ __launch_bounds__(128) void k_bn_stats(const float* __restrict__ agg,
                                                  const float* __restrict__ bgcn,
                                                  const float* __restrict__ gamma,
                                                  float* __restrict__ mean,
                                                  float* __restrict__ scale) {
    int c = threadIdx.x;
    float bc = bgcn[c];
    float s = 0.0f, s2 = 0.0f;
    for (int nn = 0; nn < NN; ++nn) {             // coalesced across c
        float v = agg[nn * HH + c] + bc;
        s += v; s2 += v * v;
    }
    float mu  = s * (1.0f / NN);
    float var = s2 * (1.0f / NN) - mu * mu;
    mean[c]  = mu;
    scale[c] = gamma[c] * rsqrtf(var + 1e-5f);
}

// ---------------- normalize + ReLU -> flattened vector v --------------------
__global__ void k_bn_relu(const float* __restrict__ agg,
                          const float* __restrict__ bgcn,
                          const float* __restrict__ beta,
                          const float* __restrict__ mean,
                          const float* __restrict__ scale,
                          float* __restrict__ v) {
    int t = blockIdx.x * blockDim.x + threadIdx.x;
    if (t < NN * HH) {
        int c = t & 127;
        float h = agg[t] + bgcn[c];
        float o = (h - mean[c]) * scale[c] + beta[c];
        v[t] = fmaxf(o, 0.0f);
    }
}

// ---------------- GEMV1: y1acc += v[slice] @ W1[slice] (split-K) ------------
// 128 blocks x 128 threads; each block owns 1024 rows of W1 (2 MB streamed).
__global__ __launch_bounds__(128) void k_gemv1(const float* __restrict__ v,
                                               const float* __restrict__ W1,
                                               float* __restrict__ y1acc) {
    __shared__ float vs[1024];
    const int t  = threadIdx.x;
    const int i0 = blockIdx.x * 1024;
#pragma unroll
    for (int r = 0; r < 8; ++r) vs[t + r * 128] = v[i0 + t + r * 128];
    __syncthreads();

    v4f acc = {0.0f, 0.0f, 0.0f, 0.0f};
    const float* Wp = W1 + (size_t)i0 * 512 + 4 * t;
    for (int i = 0; i < 1024; ++i) {
        v4f w = __builtin_nontemporal_load((const v4f*)(Wp + (size_t)i * 512));
        float vi = vs[i];
        acc[0] += vi * w[0]; acc[1] += vi * w[1];
        acc[2] += vi * w[2]; acc[3] += vi * w[3];
    }
    atomicAdd(&y1acc[4 * t + 0], acc[0]);
    atomicAdd(&y1acc[4 * t + 1], acc[1]);
    atomicAdd(&y1acc[4 * t + 2], acc[2]);
    atomicAdd(&y1acc[4 * t + 3], acc[3]);
}

// ---------------- fused ReLU(y1) + GEMV2 + ReLU -> y2 -----------------------
__global__ __launch_bounds__(256) void k_gemv2(const float* __restrict__ y1acc,
                                               const float* __restrict__ b1,
                                               const float* __restrict__ W2,
                                               const float* __restrict__ b2,
                                               float* __restrict__ y2) {
    __shared__ float y1s[512];
    const int t = threadIdx.x;
    y1s[t]       = fmaxf(y1acc[t]       + b1[t],       0.0f);
    y1s[t + 256] = fmaxf(y1acc[t + 256] + b1[t + 256], 0.0f);
    __syncthreads();
    float acc = b2[t];
    for (int j = 0; j < 512; ++j) acc += y1s[j] * W2[j * EH + t];  // coalesced
    y2[t] = fmaxf(acc, 0.0f);
}

// ---------------- GEMV3: logits = y2 @ W3 + b3 (536 MB NT stream) -----------
// 1023 blocks x 128 threads, 4 outputs/thread -> 523776 outputs exactly.
__global__ __launch_bounds__(128) void k_gemv3(const float* __restrict__ y2,
                                               const float* __restrict__ W3,
                                               const float* __restrict__ b3,
                                               float* __restrict__ out) {
    __shared__ float ys[256];
    const int t = threadIdx.x;
    ys[t]       = y2[t];
    ys[t + 128] = y2[t + 128];
    __syncthreads();

    const size_t j0 = ((size_t)blockIdx.x * 128 + t) * 4;
    v4f acc = *(const v4f*)(b3 + j0);
    for (int i = 0; i < 256; ++i) {
        v4f w = __builtin_nontemporal_load((const v4f*)(W3 + (size_t)i * MAXE + j0));
        float yi = ys[i];
        acc[0] += yi * w[0]; acc[1] += yi * w[1];
        acc[2] += yi * w[2]; acc[3] += yi * w[3];
    }
    *(v4f*)(out + j0) = acc;
}

// ---------------------------------------------------------------------------
extern "C" void kernel_launch(void* const* d_in, const int* in_sizes, int n_in,
                              void* d_out, int out_size, void* d_ws, size_t ws_size,
                              hipStream_t stream) {
    const float* x     = (const float*)d_in[0];
    const int*   ei    = (const int*)  d_in[1];
    const float* Wg    = (const float*)d_in[2];
    const float* bg    = (const float*)d_in[3];
    const float* gamma = (const float*)d_in[4];
    const float* beta  = (const float*)d_in[5];
    const float* W1    = (const float*)d_in[6];
    const float* b1    = (const float*)d_in[7];
    const float* W2    = (const float*)d_in[8];
    const float* b2    = (const float*)d_in[9];
    const float* W3    = (const float*)d_in[10];
    const float* b3    = (const float*)d_in[11];

    float* ws    = (float*)d_ws;
    float* hw    = ws + OFF_HW;    // x@W, later reused for BN output v
    float* agg   = ws + OFF_AGG;
    float* deg   = ws + OFF_DEG;   // deg -> dinv in place
    float* y1    = ws + OFF_Y1;
    float* y2    = ws + OFF_Y2;
    float* mean  = ws + OFF_MEAN;
    float* scale = ws + OFF_SCL;
    float* out   = (float*)d_out;

    // 1) zero the atomically-accumulated regions (agg, deg, y1 contiguous)
    k_zero<<<(ZERO_CNT + 255) / 256, 256, 0, stream>>>(agg, ZERO_CNT);
    // 2) GCN GEMM via WMMA (512 waves)
    k_gcn_gemm<<<64, 256, 0, stream>>>(x, Wg, hw);
    // 3) degree + dinv
    k_deg<<<(EE + NN + 255) / 256, 256, 0, stream>>>(ei, deg);
    k_dinv<<<(NN + 255) / 256, 256, 0, stream>>>(deg);
    // 4) normalized scatter-add over edges + self loops
    k_scatter<<<((EE + NN) * HH) / 256, 256, 0, stream>>>(ei, hw, deg, agg);
    // 5) BN stats, then normalize+ReLU into the flattened vector (reuse hw)
    k_bn_stats<<<1, 128, 0, stream>>>(agg, bg, gamma, mean, scale);
    k_bn_relu<<<(NN * HH) / 256, 256, 0, stream>>>(agg, bg, beta, mean, scale, hw);
    // 6) MLP head: split-K GEMV1, fused GEMV2, streaming GEMV3
    k_gemv1<<<128, 128, 0, stream>>>(hw, W1, y1);
    k_gemv2<<<1, 256, 0, stream>>>(y1, b1, W2, b2, y2);
    k_gemv3<<<MAXE / 512, 128, 0, stream>>>(y2, W3, b3, out);
}
